// SPDConv_Oct_CARAFE_HWD_17678085390696
// MI455X (gfx1250) — matrix operand order repaired and checked by reading, not compiled, not run on hardware
//
#include <hip/hip_runtime.h>
#include <hip/hip_bf16.h>

// ---------------------------------------------------------------------------
// MI455X (gfx1250, wave32) fused SPDConv + Oct + CARAFE + HWD pipeline.
// All matrix work on v_wmma_f32_16x16x32_f16 (M=16 rows, N=16 pixels, K=32).
// f16 intermediates in ws (L2-resident); CARAFE-apply + Haar fused in VALU.
// Interior global->LDS staging uses GLOBAL_LOAD_ASYNC_TO_LDS_B128 (ASYNCcnt)
// when the toolchain exposes the builtin; plain b128 copies otherwise.
// ---------------------------------------------------------------------------

typedef __attribute__((ext_vector_type(16))) _Float16 v16h;
typedef __attribute__((ext_vector_type(8)))  float    v8f;
typedef __attribute__((ext_vector_type(4)))  int      i32x4;

#define DEVINL static __device__ __forceinline__

#if __has_builtin(__builtin_amdgcn_global_load_async_to_lds_b128)
#define HAVE_ASYNC_COPY 1
#else
#define HAVE_ASYNC_COPY 0
#endif

// Per-lane 16B global->LDS copy (async when available).
// Builtin signature (from hipcc diagnostic): (i32x4 AS1*, i32x4 AS3*, Imm, Imm)
DEVINL void async_copy_b128(const _Float16* g, _Float16* l) {
#if HAVE_ASYNC_COPY
  __builtin_amdgcn_global_load_async_to_lds_b128(
      (__attribute__((address_space(1))) i32x4*)g,
      (__attribute__((address_space(3))) i32x4*)l, 0, 0);
#else
  *(uint4*)l = *(const uint4*)g;
#endif
}

DEVINL void async_join() {
#if HAVE_ASYNC_COPY
#if __has_builtin(__builtin_amdgcn_s_wait_asynccnt)
  __builtin_amdgcn_s_wait_asynccnt(0);
#else
  asm volatile("s_wait_asynccnt 0x0" ::: "memory");
#endif
#endif
}

DEVINL v8f wmma_f16(v16h a, v16h b, v8f c) {
  // (neg_a, A, neg_b, B, c_mod, C, reuse_a, reuse_b)
  return __builtin_amdgcn_wmma_f32_16x16x32_f16(false, a, false, b, (short)0, c,
                                                false, false);
}

// A fragment (16x32, row-major f16 matrix, ldK = row stride in elements).
// ISA layout: lane L holds row M=L&15; VGPR v holds K = (L>>4)*8 + (v<4 ? 2v : 16+2(v-4)) {+0,+1}.
DEVINL v16h load_a(const _Float16* W, int ldK, int mBase, int kBase, int lane) {
  const int M    = lane & 15;
  const int half = lane >> 4;
  const _Float16* rp = W + (size_t)(mBase + M) * ldK + kBase + half * 8;
  v16h a;
#pragma unroll
  for (int v = 0; v < 8; ++v) {
    const int k0 = (v < 4) ? (v * 2) : (16 + (v - 4) * 2);
    a[2 * v]     = rp[k0];
    a[2 * v + 1] = rp[k0 + 1];
  }
  return a;
}

// B fragment (32x16): pixel-major [N][K] storage; lane L holds col N=L&15,
// K = (L>>4)*8 + (e<8 ? e : 16+(e-8)), e=0..15 (same order as load_a).
DEVINL v16h load_b(const _Float16* B, int ldK, int kBase, int lane) {
  const int N = lane & 15;
  const _Float16* p = B + (size_t)N * ldK + kBase + ((lane >> 4) << 3);
  v16h b;
#pragma unroll
  for (int e = 0; e < 16; ++e) b[e] = p[e < 8 ? e : e + 8];
  return b;
}

DEVINL float silu(float v) { return v / (1.0f + __expf(-v)); }

// ---------------------------------------------------------------------------
// K0: repack weights to f16 in ws.
//   Aenc[112][576] : enc_w (100,64,3,3) -> A[m][(3di+dj)*64+c], zero-pad M.
//   Whf [64][128]  : rows 0-31 H2H_w, rows 32-63 H2L_w  (applied to hf)
//   Wlf [64][128]  : rows 0-31 L2H_w, rows 32-63 L2L_w  (applied to lf)
//   Wcomp[64][32], Whwd[32][128]
// ---------------------------------------------------------------------------
__global__ __launch_bounds__(256) void k0_prep(
    const float* __restrict__ enc_w,
    const float* __restrict__ H2H_w, const float* __restrict__ H2L_w,
    const float* __restrict__ L2H_w, const float* __restrict__ L2L_w,
    const float* __restrict__ comp_w, const float* __restrict__ hwd_w,
    _Float16* __restrict__ Aenc, _Float16* __restrict__ Whf,
    _Float16* __restrict__ Wlf, _Float16* __restrict__ Wcomp,
    _Float16* __restrict__ Whwd) {
  const int idx = blockIdx.x * 256 + threadIdx.x;
  if (idx < 64512) {                       // 112*576
    const int m = idx / 576, k = idx % 576;
    const int tap = k >> 6, c = k & 63, di = tap / 3, dj = tap % 3;
    const float v = (m < 100) ? enc_w[((m * 64 + c) * 3 + di) * 3 + dj] : 0.0f;
    Aenc[idx] = (_Float16)v;
  } else if (idx < 72704) {                // 64*128
    const int q = idx - 64512;
    const int m = q >> 7, k = q & 127;
    Whf[q] = (_Float16)((m < 32) ? H2H_w[m * 128 + k] : H2L_w[(m - 32) * 128 + k]);
  } else if (idx < 80896) {
    const int q = idx - 72704;
    const int m = q >> 7, k = q & 127;
    Wlf[q] = (_Float16)((m < 32) ? L2H_w[m * 128 + k] : L2L_w[(m - 32) * 128 + k]);
  } else if (idx < 82944) {                // 64*32
    Wcomp[idx - 80896] = (_Float16)comp_w[idx - 80896];
  } else if (idx < 87040) {                // 32*128
    Whwd[idx - 82944] = (_Float16)hwd_w[idx - 82944];
  }
}

// ---------------------------------------------------------------------------
// K1: space-to-depth + 4x conv1x1 (two 64x128 WMMA GEMMs) + comp conv (WMMA)
// block = (jhalf, i, b); 128 threads = 4 waves, 16 px/wave, 64 px tile.
// ---------------------------------------------------------------------------
__global__ __launch_bounds__(128) void k1_spd(
    const float* __restrict__ x,
    const float* __restrict__ bH2H, const float* __restrict__ bH2L,
    const float* __restrict__ bL2H, const float* __restrict__ bL2L,
    const float* __restrict__ comp_g, const float* __restrict__ comp_b,
    const _Float16* __restrict__ Whf, const _Float16* __restrict__ Wlf,
    const _Float16* __restrict__ Wcomp,
    _Float16* __restrict__ wsXc, _Float16* __restrict__ wsComp,
    float* __restrict__ wsL2H, float* __restrict__ out) {
  __shared__ __align__(16) _Float16 lds_xs[64][256];   // [pixel][xs-channel]
  __shared__ __align__(16) _Float16 lds_Xc[64][32];
  __shared__ float lds_bias[256];

  const int t = threadIdx.x;
  const int jhalf = blockIdx.x, i = blockIdx.y, b = blockIdx.z;
  const int j0 = jhalf * 64;

  if (t < 32) {
    lds_bias[t]      = bH2H[t];
    lds_bias[32 + t] = bH2L[t];
    lds_bias[64 + t] = bL2H[t];
    lds_bias[96 + t] = bL2L[t];
  }
  if (t < 64) {
    lds_bias[128 + t] = comp_g[t] * rsqrtf(1.0f + 1e-5f);  // bn scale
    lds_bias[192 + t] = comp_b[t];
  }

  // space-to-depth load: thread -> (channel c, row parity rp), streams a row.
  {
    const int c = t & 63, rp = t >> 6;
    const int y = 2 * i + rp;
    const float* xr = x + (((size_t)(b * 64 + c) * 256 + y) * 256 + 2 * j0);
    const int chE = rp ? (64 + c) : c;          // even-col quadrant channel
    const int chO = rp ? (192 + c) : (128 + c); // odd-col quadrant channel
#pragma unroll 4
    for (int p2 = 0; p2 < 32; ++p2) {
      const float4 v = *(const float4*)(xr + 4 * p2);
      lds_xs[2 * p2][chE]     = (_Float16)v.x;
      lds_xs[2 * p2][chO]     = (_Float16)v.y;
      lds_xs[2 * p2 + 1][chE] = (_Float16)v.z;
      lds_xs[2 * p2 + 1][chO] = (_Float16)v.w;
    }
  }
  __syncthreads();

  const int lane = t & 31, wave = t >> 5;
  const int p0 = wave * 16, N = lane & 15, rowoff = (lane >> 4) * 8;
  const int p = p0 + N, j = j0 + p;

  v8f accH[4] = {{0}}, accL[4] = {{0}};
#pragma unroll
  for (int kk = 0; kk < 4; ++kk) {
    const v16h bf = load_b(&lds_xs[p0][0], 256, kk * 32, lane);   // hf channels
#pragma unroll
    for (int m = 0; m < 4; ++m)
      accH[m] = wmma_f16(load_a(Whf, 128, m * 16, kk * 32, lane), bf, accH[m]);
  }
#pragma unroll
  for (int kk = 0; kk < 4; ++kk) {
    const v16h bf = load_b(&lds_xs[p0][0], 256, 128 + kk * 32, lane);  // lf
#pragma unroll
    for (int m = 0; m < 4; ++m)
      accL[m] = wmma_f16(load_a(Wlf, 128, m * 16, kk * 32, lane), bf, accL[m]);
  }

  const size_t pix = (size_t)(b * 128 + i) * 128 + j;
#pragma unroll
  for (int m = 0; m < 2; ++m)
#pragma unroll
    for (int r = 0; r < 8; ++r) {
      const int ch = m * 16 + r + rowoff;                 // 0..31
      const float xc = accH[m][r] + lds_bias[ch];         // h2h pre-CARAFE
      lds_Xc[p][ch]        = (_Float16)xc;
      wsXc[pix * 32 + ch]  = (_Float16)xc;
      wsL2H[pix * 32 + ch] = accL[m][r] + lds_bias[64 + ch];  // l2h
      // lf_ = h2l + l2l (+ both biases) -> out channels 32..63
      out[(((size_t)(b * 64 + 32 + ch)) * 128 + i) * 128 + j] =
          accH[m + 2][r] + accL[m + 2][r] + lds_bias[32 + ch] + lds_bias[96 + ch];
    }
  __syncthreads();

  // comp conv: [64x16] = Wcomp[64x32] * Xc[32x16], then BN + SiLU
  v8f accC[4] = {{0}};
  {
    const v16h bf = load_b(&lds_Xc[p0][0], 32, 0, lane);
#pragma unroll
    for (int m = 0; m < 4; ++m)
      accC[m] = wmma_f16(load_a(Wcomp, 32, m * 16, 0, lane), bf, accC[m]);
  }
#pragma unroll
  for (int m = 0; m < 4; ++m)
#pragma unroll
    for (int r = 0; r < 8; ++r) {
      const int ch = m * 16 + r + rowoff;                 // 0..63
      const float v = accC[m][r] * lds_bias[128 + ch] + lds_bias[192 + ch];
      wsComp[pix * 64 + ch] = (_Float16)silu(v);
    }
}

// ---------------------------------------------------------------------------
// K2: enc 3x3 conv 64->100 as implicit GEMM (9 taps x 2 ksteps x 7 Mtiles
// WMMAs per wave), BN, then softmax over 25 per (pixel, subpixel).
// ---------------------------------------------------------------------------
__global__ __launch_bounds__(128) void k2_enc(
    const _Float16* __restrict__ wsComp,
    const _Float16* __restrict__ Aenc,
    const float* __restrict__ enc_g, const float* __restrict__ enc_b,
    _Float16* __restrict__ wsWsm) {
  __shared__ __align__(16) _Float16 lds_in[3][66][64];  // rows i-1..i+1, cols j0-1..j0+64
  __shared__ float lds_P[64][104];
  __shared__ float lds_s[112], lds_bb[112];

  const int t = threadIdx.x;
  const int jhalf = blockIdx.x, i = blockIdx.y, b = blockIdx.z;
  const int j0 = jhalf * 64;

  if (t < 112) {
    lds_s[t]  = (t < 100) ? enc_g[t] * rsqrtf(1.0f + 1e-5f) : 0.0f;
    lds_bb[t] = (t < 100) ? enc_b[t] : 0.0f;
  }
  // halo columns + out-of-range rows: scalar path with zero fill
  for (int idx = t; idx < 3 * 66 * 64; idx += 128) {
    const int rr = idx / (66 * 64);
    const int rem = idx - rr * 66 * 64;
    const int cc = rem >> 6, c = rem & 63;
    const int ii = i - 1 + rr, j = j0 - 1 + cc;
    if ((unsigned)ii < 128u && cc >= 1 && cc <= 64) continue;  // interior below
    _Float16 v = (_Float16)0.0f;
    if ((unsigned)ii < 128u && (unsigned)j < 128u)
      v = wsComp[((size_t)(b * 128 + ii) * 128 + j) * 64 + c];
    lds_in[rr][cc][c] = v;
  }
  // interior rows: 64 px x 64 ch per row, 16B per lane, async to LDS
  for (int rr = 0; rr < 3; ++rr) {
    const int ii = i - 1 + rr;             // block-uniform predicate
    if ((unsigned)ii >= 128u) continue;
    const _Float16* grow = wsComp + ((size_t)(b * 128 + ii) * 128 + j0) * 64;
    for (int chunk = t; chunk < 512; chunk += 128) {
      const int pp = chunk >> 3, c8 = (chunk & 7) << 3;
      async_copy_b128(grow + pp * 64 + c8, &lds_in[rr][pp + 1][c8]);
    }
  }
  async_join();
  __syncthreads();

  const int lane = t & 31, wave = t >> 5;
  const int p0 = wave * 16, N = lane & 15, rowoff = (lane >> 4) * 8;
  const int p = p0 + N;

  v8f acc[7] = {{0}};
#pragma unroll
  for (int tap = 0; tap < 9; ++tap) {
    const int di = tap / 3, dj = tap % 3;
#pragma unroll
    for (int ks = 0; ks < 2; ++ks) {
      const v16h bf = load_b(&lds_in[di][p0 + dj][0], 64, ks * 32, lane);
#pragma unroll
      for (int m = 0; m < 7; ++m)
        acc[m] = wmma_f16(load_a(Aenc, 576, m * 16, tap * 64 + ks * 32, lane),
                          bf, acc[m]);
    }
  }
#pragma unroll
  for (int m = 0; m < 7; ++m)
#pragma unroll
    for (int r = 0; r < 8; ++r) {
      const int ch = m * 16 + r + rowoff;
      if (ch < 100) lds_P[p][ch] = acc[m][r] * lds_s[ch] + lds_bb[ch];
    }
  __syncthreads();

  // softmax over k=25 for each (pixel, s=s1*2+s2); pixel-shuffle: ch = 4k+s
  for (int item = t; item < 256; item += 128) {
    const int pp = item >> 2, s = item & 3;
    float mx = -1e30f;
#pragma unroll
    for (int k = 0; k < 25; ++k) mx = fmaxf(mx, lds_P[pp][4 * k + s]);
    float e[25], sum = 0.0f;
#pragma unroll
    for (int k = 0; k < 25; ++k) { e[k] = __expf(lds_P[pp][4 * k + s] - mx); sum += e[k]; }
    const float inv = 1.0f / sum;
    _Float16* wp = wsWsm + ((size_t)(b * 128 + i) * 128 + j0 + pp) * 100 + s * 25;
#pragma unroll
    for (int k = 0; k < 25; ++k) wp[k] = (_Float16)(e[k] * inv);
  }
}

// ---------------------------------------------------------------------------
// K3: CARAFE apply (+l2h) fused with Haar downsample (hf_ never hits HBM),
// then hwd conv 128->32 via WMMA + BN + SiLU -> out channels 0..31.
// ---------------------------------------------------------------------------
__global__ __launch_bounds__(128) void k3_carafe_hwd(
    const _Float16* __restrict__ wsXc,
    const _Float16* __restrict__ wsWsm,
    const float* __restrict__ wsL2H,
    const _Float16* __restrict__ Whwd,
    const float* __restrict__ hwd_g, const float* __restrict__ hwd_b,
    float* __restrict__ out) {
  __shared__ __align__(16) _Float16 lds_X[5][68][32];  // rows i-2..i+2, cols j0-2..j0+65
  __shared__ __align__(16) _Float16 lds_W[64][100];
  __shared__ __align__(16) float lds_l2h[64][32];
  __shared__ __align__(16) _Float16 lds_y[64][128];
  __shared__ float lds_s[32], lds_bb[32];

  const int t = threadIdx.x;
  const int jhalf = blockIdx.x, i = blockIdx.y, b = blockIdx.z;
  const int j0 = jhalf * 64;

  if (t < 32) { lds_s[t] = hwd_g[t] * rsqrtf(1.0f + 1e-5f); lds_bb[t] = hwd_b[t]; }

  // halo / OOR elements of X (scalar, zero fill)
  for (int idx = t; idx < 5 * 68 * 32; idx += 128) {
    const int rr = idx / (68 * 32);
    const int rem = idx - rr * 68 * 32;
    const int cc = rem >> 5, c = rem & 31;
    const int ii = i - 2 + rr, j = j0 - 2 + cc;
    if ((unsigned)ii < 128u && cc >= 2 && cc <= 65) continue;  // interior below
    _Float16 v = (_Float16)0.0f;
    if ((unsigned)ii < 128u && (unsigned)j < 128u)
      v = wsXc[((size_t)(b * 128 + ii) * 128 + j) * 32 + c];
    lds_X[rr][cc][c] = v;
  }
  // interior rows of X: 64 px x 32 ch per row, async b128
  for (int rr = 0; rr < 5; ++rr) {
    const int ii = i - 2 + rr;             // block-uniform predicate
    if ((unsigned)ii >= 128u) continue;
    const _Float16* grow = wsXc + ((size_t)(b * 128 + ii) * 128 + j0) * 32;
    for (int chunk = t; chunk < 256; chunk += 128) {
      const int pp = chunk >> 2, c8 = (chunk & 3) << 3;
      async_copy_b128(grow + pp * 32 + c8, &lds_X[rr][pp + 2][c8]);
    }
  }
  // softmax weights: 64 px x 100 ch, 8B chunks (100 % 4 == 0)
  {
    const _Float16* wrow = wsWsm + ((size_t)(b * 128 + i) * 128 + j0) * 100;
    for (int chunk = t; chunk < 1600; chunk += 128) {
      const int pp = chunk / 25, q = (chunk % 25) * 4;
      *(uint2*)&lds_W[pp][q] = *(const uint2*)(wrow + pp * 100 + q);
    }
  }
  // l2h: 64 px x 32 f32, 16B chunks
  {
    const float* lrow = wsL2H + ((size_t)(b * 128 + i) * 128 + j0) * 32;
    for (int chunk = t; chunk < 512; chunk += 128) {
      const int pp = chunk >> 3, c4 = (chunk & 7) << 2;
      *(float4*)&lds_l2h[pp][c4] = *(const float4*)(lrow + pp * 32 + c4);
    }
  }
  async_join();
  __syncthreads();

  // CARAFE 5x5 apply for the 4 subpixels + Haar combine -> y[128] in LDS
  {
    const int pp = t >> 1, cb = (t & 1) * 16;
    float a0[16], a1[16], a2[16], a3[16];
#pragma unroll
    for (int c = 0; c < 16; ++c) {
      const float l = lds_l2h[pp][cb + c];   // l2h nearest-upsample (same for 4)
      a0[c] = l; a1[c] = l; a2[c] = l; a3[c] = l;
    }
    for (int tap = 0; tap < 25; ++tap) {
      const int di = tap / 5, dj = tap % 5;
      const float w0 = (float)lds_W[pp][tap];        // s=(0,0) = a
      const float w1 = (float)lds_W[pp][25 + tap];   // s=(0,1) = b
      const float w2 = (float)lds_W[pp][50 + tap];   // s=(1,0) = c
      const float w3 = (float)lds_W[pp][75 + tap];   // s=(1,1) = d
      const _Float16* xp = &lds_X[di][pp + dj][cb];
#pragma unroll
      for (int c = 0; c < 16; ++c) {
        const float xv = (float)xp[c];
        a0[c] += w0 * xv; a1[c] += w1 * xv; a2[c] += w2 * xv; a3[c] += w3 * xv;
      }
    }
#pragma unroll
    for (int c = 0; c < 16; ++c) {
      const float a = a0[c], bb2 = a1[c], cc2 = a2[c], dd = a3[c];
      lds_y[pp][cb + c]       = (_Float16)((a + bb2 + cc2 + dd) * 0.5f);  // ll
      lds_y[pp][32 + cb + c]  = (_Float16)((a - bb2 + cc2 - dd) * 0.5f);  // lh
      lds_y[pp][64 + cb + c]  = (_Float16)((a + bb2 - cc2 - dd) * 0.5f);  // hl
      lds_y[pp][96 + cb + c]  = (_Float16)((a - bb2 - cc2 + dd) * 0.5f);  // hh
    }
  }
  __syncthreads();

  const int lane = t & 31, wave = t >> 5;
  const int p0 = wave * 16, N = lane & 15, rowoff = (lane >> 4) * 8;
  v8f acc[2] = {{0}};
#pragma unroll
  for (int kk = 0; kk < 4; ++kk) {
    const v16h bf = load_b(&lds_y[p0][0], 128, kk * 32, lane);
#pragma unroll
    for (int m = 0; m < 2; ++m)
      acc[m] = wmma_f16(load_a(Whwd, 128, m * 16, kk * 32, lane), bf, acc[m]);
  }
  const int j = j0 + p0 + N;
#pragma unroll
  for (int m = 0; m < 2; ++m)
#pragma unroll
    for (int r = 0; r < 8; ++r) {
      const int o = m * 16 + r + rowoff;
      const float v = acc[m][r] * lds_s[o] + lds_bb[o];
      out[(((size_t)(b * 64 + o)) * 128 + i) * 128 + j] = silu(v);
    }
}

// ---------------------------------------------------------------------------
extern "C" void kernel_launch(void* const* d_in, const int* in_sizes, int n_in,
                              void* d_out, int out_size, void* d_ws, size_t ws_size,
                              hipStream_t stream) {
  const float* x      = (const float*)d_in[0];
  const float* L2L_w  = (const float*)d_in[1];
  const float* L2L_b  = (const float*)d_in[2];
  const float* L2H_w  = (const float*)d_in[3];
  const float* L2H_b  = (const float*)d_in[4];
  const float* H2L_w  = (const float*)d_in[5];
  const float* H2L_b  = (const float*)d_in[6];
  const float* H2H_w  = (const float*)d_in[7];
  const float* H2H_b  = (const float*)d_in[8];
  const float* comp_w = (const float*)d_in[9];
  const float* comp_g = (const float*)d_in[10];
  const float* comp_b = (const float*)d_in[11];
  const float* enc_w  = (const float*)d_in[12];
  const float* enc_g  = (const float*)d_in[13];
  const float* enc_b  = (const float*)d_in[14];
  const float* hwd_w  = (const float*)d_in[15];
  const float* hwd_g  = (const float*)d_in[16];
  const float* hwd_b  = (const float*)d_in[17];
  float* out = (float*)d_out;

  // ws layout (bytes); total ~34.3 MB
  char* ws = (char*)d_ws;
  _Float16* Aenc   = (_Float16*)(ws + 0x00000);                 // 129,024
  _Float16* Whf    = (_Float16*)(ws + 0x20000);                 // 16,384
  _Float16* Wlf    = (_Float16*)(ws + 0x24000);                 // 16,384
  _Float16* Wcomp  = (_Float16*)(ws + 0x28000);                 // 4,096
  _Float16* Whwd   = (_Float16*)(ws + 0x29000);                 // 8,192
  size_t o = 0x2C000;
  _Float16* wsXc   = (_Float16*)(ws + o);  o += 4u * 128 * 128 * 32 * 2;   // 4.19 MB
  _Float16* wsComp = (_Float16*)(ws + o);  o += 4u * 128 * 128 * 64 * 2;   // 8.39 MB
  float*    wsL2H  = (float*)   (ws + o);  o += 4u * 128 * 128 * 32 * 4;   // 8.39 MB
  _Float16* wsWsm  = (_Float16*)(ws + o);                                  // 13.1 MB

  k0_prep<<<340, 256, 0, stream>>>(enc_w, H2H_w, H2L_w, L2H_w, L2L_w, comp_w,
                                   hwd_w, Aenc, Whf, Wlf, Wcomp, Whwd);

  dim3 grid(2, 128, 4);  // (jhalf, row, batch)
  k1_spd<<<grid, 128, 0, stream>>>(x, H2H_b, H2L_b, L2H_b, L2L_b, comp_g, comp_b,
                                   Whf, Wlf, Wcomp, wsXc, wsComp, wsL2H, out);
  k2_enc<<<grid, 128, 0, stream>>>(wsComp, Aenc, enc_g, enc_b, wsWsm);
  k3_carafe_hwd<<<grid, 128, 0, stream>>>(wsXc, wsWsm, wsL2H, Whwd, hwd_g, hwd_b,
                                          out);
  (void)in_sizes; (void)n_in; (void)out_size; (void)ws_size;
}
